// SymbolicConstrainedTransformer_11338713661448
// MI455X (gfx1250) — matrix operand order repaired and checked by reading, not compiled, not run on hardware
//
#include <hip/hip_runtime.h>
#include <hip/hip_bf16.h>
#include <math.h>

// ---------------- problem constants (from reference) ----------------
constexpr int Vv  = 32000;
constexpr int Dd  = 1024;
constexpr int Hh  = 16;
constexpr int NLn = 4;
constexpr int FFf = 4096;
constexpr int Bb  = 2;
constexpr int Ll  = 2048;
constexpr int DHh = 64;
constexpr int Mrows = Bb * Ll;  // 4096

// ---------------- vector types ----------------
typedef __attribute__((ext_vector_type(16))) _Float16 v16h;
typedef __attribute__((ext_vector_type(8)))  _Float16 v8h;
typedef __attribute__((ext_vector_type(8)))  float    v8f;
typedef __attribute__((ext_vector_type(4)))  float    v4f;

__device__ __forceinline__ v8f wmma_f16(v16h a, v16h b, v8f c) {
  // v_wmma_f32_16x16x32_f16 : D = A(16x32 f16) * B(32x16 f16) + C(16x16 f32)
  return __builtin_amdgcn_wmma_f32_16x16x32_f16(
      false, a, false, b, (short)0, c, false, false);
}

// A-fragment (16x32, lane m = lane&15, kb = lane>=16 ? 16 : 0) built from a
// row-major f32 row pointer: elements = p[kb..kb+15].
// Works for global f32 (global_load_b128 x4) and LDS f32 (ds_load_b128 x4).
__device__ __forceinline__ v16h a_frag_f32(const float* p, int kb) {
  const v4f r0 = *(const v4f*)(p + kb);
  const v4f r1 = *(const v4f*)(p + kb + 4);
  const v4f r2 = *(const v4f*)(p + kb + 8);
  const v4f r3 = *(const v4f*)(p + kb + 12);
  v16h f;
#pragma unroll
  for (int e = 0; e < 4; ++e) {
    f[e]      = (_Float16)r0[e];
    f[4 + e]  = (_Float16)r1[e];
    f[8 + e]  = (_Float16)r2[e];
    f[12 + e] = (_Float16)r3[e];
  }
  return f;
}

// B-fragment from 16 *contiguous* f32 values (used for Q*K^T: a K-row is
// contiguous along dh, which is exactly B[k][n] with k consecutive per lane).
__device__ __forceinline__ v16h b_frag16_f32(const float* p) {
  const v4f r0 = *(const v4f*)(p);
  const v4f r1 = *(const v4f*)(p + 4);
  const v4f r2 = *(const v4f*)(p + 8);
  const v4f r3 = *(const v4f*)(p + 12);
  v16h f;
#pragma unroll
  for (int e = 0; e < 4; ++e) {
    f[e]      = (_Float16)r0[e];
    f[4 + e]  = (_Float16)r1[e];
    f[8 + e]  = (_Float16)r2[e];
    f[12 + e] = (_Float16)r3[e];
  }
  return f;
}

// ---------------- kernel 1: embedding + sinusoidal PE ----------------
__global__ __launch_bounds__(256) void embed_pe_kernel(
    const long long* __restrict__ ids, const float* __restrict__ emb,
    float* __restrict__ x) {
  const int bl = blockIdx.x;          // 0 .. B*L-1
  const int l  = bl % Ll;
  const long long id = ids[bl];
  const float* er = emb + (size_t)id * Dd;
  float* xr = x + (size_t)bl * Dd;
  const float negln = -9.210340371976184f; // -ln(10000)
  for (int d = threadIdx.x; d < Dd; d += 256) {
    const int i = d >> 1;
    const float ang = (float)l * __expf(negln * (2.0f * (float)i / (float)Dd));
    const float pe  = (d & 1) ? __cosf(ang) : __sinf(ang);
    xr[d] = er[d] + pe;
  }
}

// ---------------- kernel 2: weight transpose prepass ----------------
// W: [K][N] f32 row-major  ->  WT: [N][K] f16 row-major.
// 32x32 LDS tile: coalesced f32 reads, vectorized f16 writes.
__global__ __launch_bounds__(256) void wtrans_kernel(
    const float* __restrict__ W, _Float16* __restrict__ WT, int K, int N) {
  constexpr int TLD = 40;             // 32 + pad (keeps 16B-aligned rows)
  __shared__ _Float16 T[32 * TLD];
  const int tid = threadIdx.x;
  const int k0 = blockIdx.y * 32;
  const int n0 = blockIdx.x * 32;
#pragma unroll
  for (int i = 0; i < 4; ++i) {
    const int idx = i * 256 + tid;
    const int r = idx >> 5, c = idx & 31;        // r: k, c: n
    T[c * TLD + r] = (_Float16)W[(size_t)(k0 + r) * N + (n0 + c)];
  }
  __syncthreads();
  if (tid < 128) {
    const int rr = tid >> 2;           // n within tile
    const int cc = (tid & 3) * 8;      // k chunk
    *(v8h*)(WT + (size_t)(n0 + rr) * K + k0 + cc) = *(const v8h*)(T + rr * TLD + cc);
  }
}

// ---------------- kernel 3: WMMA GEMM, LDS-free ----------------
// C[M,N] = A[M,K](f32) @ B  with B given transposed as BT[N][K] f16.
// 256 threads = 8 wave32s, each wave owns a 16x64 output tile. All fragments
// are contiguous loads (64B f32 A, 32B f16 B) straight from global (L2-resident on MI455X).
// Optional: ReLU; optional f16 transposed tap of columns >= 2D into vT
// (builds V^T [B][D][L] during the QKV GEMM for the attention kernel).
__global__ __launch_bounds__(256) void gemm_at_kernel(
    const float* __restrict__ A, const _Float16* __restrict__ BT,
    const float* __restrict__ bias, float* __restrict__ C,
    int M, int N, int K, int relu, _Float16* __restrict__ vT) {
  const int tid  = threadIdx.x;
  const int lane = tid & 31;
  const int w    = tid >> 5;                 // wave 0..7 -> 16-row slice
  const int bm   = blockIdx.y * 128;
  const int bn   = blockIdx.x * 64;
  const int n    = lane & 15;
  const int kbA  = (lane & 16) ? 16 : 0;
  const int kbB  = (lane & 16) ? 16 : 0;

  const float* arow = A + (size_t)(bm + w * 16 + (lane & 15)) * K;
  const _Float16* bt0 = BT + (size_t)(bn +  0 + n) * K + kbB;
  const _Float16* bt1 = BT + (size_t)(bn + 16 + n) * K + kbB;
  const _Float16* bt2 = BT + (size_t)(bn + 32 + n) * K + kbB;
  const _Float16* bt3 = BT + (size_t)(bn + 48 + n) * K + kbB;

  v8f acc0 = {}, acc1 = {}, acc2 = {}, acc3 = {};

  for (int k0 = 0; k0 < K; k0 += 32) {
    if (k0 + 32 < K) {
      __builtin_prefetch(bt0 + k0 + 32, 0, 1);   // global_prefetch_b8
      __builtin_prefetch(arow + k0 + 32, 0, 1);
    }
    const v16h af  = a_frag_f32(arow + k0, kbA);
    const v16h bf0 = *(const v16h*)(bt0 + k0);
    const v16h bf1 = *(const v16h*)(bt1 + k0);
    const v16h bf2 = *(const v16h*)(bt2 + k0);
    const v16h bf3 = *(const v16h*)(bt3 + k0);
    acc0 = wmma_f16(af, bf0, acc0);
    acc1 = wmma_f16(af, bf1, acc1);
    acc2 = wmma_f16(af, bf2, acc2);
    acc3 = wmma_f16(af, bf3, acc3);
  }

  // epilogue: C/D layout -> lane col n, rows v + 8*(lane>=16)
  const int mb = (lane & 16) ? 8 : 0;
#pragma unroll
  for (int nt = 0; nt < 4; ++nt) {
    const v8f acc = (nt == 0) ? acc0 : (nt == 1) ? acc1 : (nt == 2) ? acc2 : acc3;
    const int col = bn + nt * 16 + n;
    const float bv = bias[col];
#pragma unroll
    for (int v = 0; v < 8; ++v) {
      const int row = bm + w * 16 + v + mb;
      float val = acc[v] + bv;
      if (relu) val = fmaxf(val, 0.0f);
      C[(size_t)row * N + col] = val;
      if (vT && col >= 2 * Dd) {   // transposed f16 tap of the V block
        const int b = row >> 11;               // / Ll
        const int l = row & (Ll - 1);
        vT[((size_t)b * Dd + (col - 2 * Dd)) * Ll + l] = (_Float16)val;
      }
    }
  }
}

// ---------------- kernel 4: causal flash attention ----------------
// One wave per 16 query rows of one (b,h). Q and K^T fragments are direct
// contiguous global f32 loads; V fragments are contiguous v16h loads from the
// pre-transposed vT; only the 16x32 softmax tile lives in LDS.
__global__ __launch_bounds__(32) void attn_kernel(
    const float* __restrict__ qkv, const _Float16* __restrict__ vT,
    float* __restrict__ out) {
  const int lane = threadIdx.x;
  const int qt = blockIdx.x;
  const int h  = blockIdx.y;
  const int bz = blockIdx.z;
  const int m   = lane & 15;
  const int n   = lane & 15;
  const int kbA = (lane & 16) ? 16 : 0;
  const int kbB = (lane & 16) ? 16 : 0;
  const int mb  = (lane & 16) ? 8 : 0;

  const size_t base = (size_t)bz * Ll * (3 * Dd);
  __shared__ float Sf[16 * 32];
  __shared__ float m_i[16], l_i[16], rowscale[16];

  // Q fragments straight from global (dh 0..31 and 32..63)
  const float* qrow = qkv + base + (size_t)(qt * 16 + m) * (3 * Dd) + h * DHh;
  const v16h aQ0 = a_frag_f32(qrow, kbA);
  const v16h aQ1 = a_frag_f32(qrow + 32, kbA);

  if (lane < 16) { m_i[lane] = -3.0e38f; l_i[lane] = 0.0f; }
  __syncthreads();

  v8f o0 = {}, o1 = {}, o2 = {}, o3 = {};
  const int qhi = qt * 16 + 15;
  const _Float16* vcol = vT + ((size_t)bz * Dd + h * DHh + n) * Ll;

  for (int kt = 0; kt * 32 <= qhi; ++kt) {
    // ---- S = Q * K^T : B[k=dh][n=key]; a K-row is contiguous along dh ----
    v8f s0 = {}, s1 = {};
    {
      const float* kp0 = qkv + base + (size_t)(kt * 32 + n) * (3 * Dd) + Dd + h * DHh + kbB;
      const float* kp1 = kp0 + (size_t)16 * (3 * Dd);
      s0 = wmma_f16(aQ0, b_frag16_f32(kp0), s0);
      s0 = wmma_f16(aQ1, b_frag16_f32(kp0 + 32), s0);
      s1 = wmma_f16(aQ0, b_frag16_f32(kp1), s1);
      s1 = wmma_f16(aQ1, b_frag16_f32(kp1 + 32), s1);
    }

    // scale + causal mask, spill 16x32 tile to LDS
#pragma unroll
    for (int t = 0; t < 2; ++t) {
      const v8f s = t ? s1 : s0;
#pragma unroll
      for (int v = 0; v < 8; ++v) {
        const int mm = v + mb;
        const int col = t * 16 + n;
        float val = s[v] * 0.125f;                  // 1/sqrt(64)
        if (kt * 32 + col > qt * 16 + mm) val = -1.0e9f;
        Sf[mm * 32 + col] = val;
      }
    }
    __syncthreads();

    // online softmax (16 lanes, one per query row)
    if (lane < 16) {
      const int r = lane;
      float tm = -3.0e38f;
      for (int c = 0; c < 32; ++c) tm = fmaxf(tm, Sf[r * 32 + c]);
      const float nm = fmaxf(m_i[r], tm);
      const float sc = __expf(m_i[r] - nm);
      rowscale[r] = sc;
      float sum = 0.0f;
      for (int c = 0; c < 32; ++c) {
        const float p = __expf(Sf[r * 32 + c] - nm);
        Sf[r * 32 + c] = p;
        sum += p;
      }
      l_i[r] = l_i[r] * sc + sum;
      m_i[r] = nm;
    }
    __syncthreads();

    // rescale running output; O += P * V
#pragma unroll
    for (int v = 0; v < 8; ++v) {
      const float sc = rowscale[v + mb];
      o0[v] *= sc; o1[v] *= sc; o2[v] *= sc; o3[v] *= sc;
    }
    const v16h aP = a_frag_f32(Sf + m * 32, kbA);         // LDS, ds_load_b128 x4
    const v16h bv0 = *(const v16h*)(vcol + (size_t) 0 * Ll + kt * 32 + kbB);
    const v16h bv1 = *(const v16h*)(vcol + (size_t)16 * Ll + kt * 32 + kbB);
    const v16h bv2 = *(const v16h*)(vcol + (size_t)32 * Ll + kt * 32 + kbB);
    const v16h bv3 = *(const v16h*)(vcol + (size_t)48 * Ll + kt * 32 + kbB);
    o0 = wmma_f16(aP, bv0, o0);
    o1 = wmma_f16(aP, bv1, o1);
    o2 = wmma_f16(aP, bv2, o2);
    o3 = wmma_f16(aP, bv3, o3);
    __syncthreads();
  }

  // normalize by row sums and scatter to [B,L,D] at head offset
#pragma unroll
  for (int v = 0; v < 8; ++v) {
    const int mm = v + mb;
    const float inv = 1.0f / l_i[mm];
    float* orow = out + ((size_t)bz * Ll + qt * 16 + mm) * Dd + h * DHh;
    orow[0 * 16 + n] = o0[v] * inv;
    orow[1 * 16 + n] = o1[v] * inv;
    orow[2 * 16 + n] = o2[v] * inv;
    orow[3 * 16 + n] = o3[v] * inv;
  }
}

// ---------------- kernel 5: y = LayerNorm(x + r) * g + b ----------------
__global__ __launch_bounds__(256) void add_ln_kernel(
    const float* __restrict__ x, const float* __restrict__ r,
    const float* __restrict__ g, const float* __restrict__ bta,
    float* __restrict__ y) {
  const int row = blockIdx.x;
  const int tid = threadIdx.x;
  const float* xr = x + (size_t)row * Dd;
  const float* rr = r + (size_t)row * Dd;
  float vals[4];
  float sum = 0.0f, sq = 0.0f;
#pragma unroll
  for (int i = 0; i < 4; ++i) {
    const int d = tid + i * 256;
    const float v = xr[d] + rr[d];
    vals[i] = v; sum += v; sq += v * v;
  }
  __shared__ float s1[256], s2[256];
  s1[tid] = sum; s2[tid] = sq;
  __syncthreads();
  for (int st = 128; st > 0; st >>= 1) {
    if (tid < st) { s1[tid] += s1[tid + st]; s2[tid] += s2[tid + st]; }
    __syncthreads();
  }
  const float mu  = s1[0] / (float)Dd;
  const float var = s2[0] / (float)Dd - mu * mu;
  const float rstd = rsqrtf(var + 1e-5f);
  float* yr = y + (size_t)row * Dd;
#pragma unroll
  for (int i = 0; i < 4; ++i) {
    const int d = tid + i * 256;
    yr[d] = (vals[i] - mu) * rstd * g[d] + bta[d];
  }
}

// ---------------- kernel 6: relation scores ----------------
// Every softmax row sums to 1 => head_patterns[b,h] == 1/L exactly.
__global__ void rel_scores_kernel(const float* __restrict__ w,
                                  const float* __restrict__ bias,
                                  float* __restrict__ out, float invL) {
  const int i = threadIdx.x;
  if (i < Bb * 4) {
    const int nrel = i & 3;
    float s = bias[nrel];
    for (int hh = 0; hh < Hh; ++hh) s += w[hh * 4 + nrel] * invL;
    out[i] = s;
  }
}

// ---------------- host orchestration ----------------
extern "C" void kernel_launch(void* const* d_in, const int* in_sizes, int n_in,
                              void* d_out, int out_size, void* d_ws, size_t ws_size,
                              hipStream_t stream) {
  (void)in_sizes; (void)n_in; (void)out_size; (void)ws_size;

  const long long* ids   = (const long long*)d_in[0];
  const float* embed     = (const float*)d_in[1];
  const float* enc_qkv_w = (const float*)d_in[2];
  const float* enc_qkv_b = (const float*)d_in[3];
  const float* enc_out_w = (const float*)d_in[4];
  const float* enc_out_b = (const float*)d_in[5];
  const float* enc_ln1_g = (const float*)d_in[6];
  const float* enc_ln1_b = (const float*)d_in[7];
  const float* enc_ln2_g = (const float*)d_in[8];
  const float* enc_ln2_b = (const float*)d_in[9];
  const float* enc_ff1_w = (const float*)d_in[10];
  const float* enc_ff1_b = (const float*)d_in[11];
  const float* enc_ff2_w = (const float*)d_in[12];
  const float* enc_ff2_b = (const float*)d_in[13];
  const float* rel_qkv_w = (const float*)d_in[14];
  const float* rel_qkv_b = (const float*)d_in[15];
  const float* rel_out_w = (const float*)d_in[16];
  const float* rel_out_b = (const float*)d_in[17];
  const float* rel_cls_w = (const float*)d_in[18];
  const float* rel_cls_b = (const float*)d_in[19];
  const float* proj_w    = (const float*)d_in[20];
  const float* proj_b    = (const float*)d_in[21];

  // ---- f32 scratch ----
  float* ws   = (float*)d_ws;
  float* x    = ws;                                 // [M, D]
  float* qkv  = x    + (size_t)Mrows * Dd;          // [M, 3D]
  float* attn = qkv  + (size_t)Mrows * 3 * Dd;      // [M, D]
  float* hbuf = attn + (size_t)Mrows * Dd;          // [M, FF]
  float* tmp2 = hbuf + (size_t)Mrows * FFf;         // [M, D]

  // ---- f16 scratch: transposed weights + transposed V ----
  _Float16* h16   = (_Float16*)(tmp2 + (size_t)Mrows * Dd);
  _Float16* vT    = h16;                                     // [B, D, L]
  _Float16* wqkv  = vT   + (size_t)Bb * Dd * Ll;             // NL x [3D, D]
  _Float16* wout  = wqkv + (size_t)NLn * Dd * 3 * Dd;        // NL x [D, D]
  _Float16* wff1  = wout + (size_t)NLn * Dd * Dd;            // NL x [FF, D]
  _Float16* wff2  = wff1 + (size_t)NLn * Dd * FFf;           // NL x [D, FF]
  _Float16* wrqkv = wff2 + (size_t)NLn * FFf * Dd;           // [3D, D]
  _Float16* wrout = wrqkv + (size_t)Dd * 3 * Dd;             // [D, D]
  _Float16* wproj = wrout + (size_t)Dd * Dd;                 // [V, D]

  // ---- prepass: transpose + f16-convert all weights ----
  auto tr = [&](const float* W, _Float16* WT, int K, int N) {
    wtrans_kernel<<<dim3(N / 32, K / 32), 256, 0, stream>>>(W, WT, K, N);
  };
  for (int li = 0; li < NLn; ++li) {
    tr(enc_qkv_w + (size_t)li * Dd * 3 * Dd, wqkv + (size_t)li * Dd * 3 * Dd, Dd, 3 * Dd);
    tr(enc_out_w + (size_t)li * Dd * Dd,     wout + (size_t)li * Dd * Dd,     Dd, Dd);
    tr(enc_ff1_w + (size_t)li * Dd * FFf,    wff1 + (size_t)li * Dd * FFf,    Dd, FFf);
    tr(enc_ff2_w + (size_t)li * FFf * Dd,    wff2 + (size_t)li * FFf * Dd,    FFf, Dd);
  }
  tr(rel_qkv_w, wrqkv, Dd, 3 * Dd);
  tr(rel_out_w, wrout, Dd, Dd);
  tr(proj_w,    wproj, Dd, Vv);

  const dim3 gemmBlk(256);
  const dim3 attnGrid(Ll / 16, Hh, Bb);

  // x = embed[ids] + PE
  embed_pe_kernel<<<Mrows, 256, 0, stream>>>(ids, embed, x);

  for (int li = 0; li < NLn; ++li) {
    // qkv = x @ qkv_w + qkv_b   (+ f16 V^T tap)
    gemm_at_kernel<<<dim3(3 * Dd / 64, Mrows / 128), gemmBlk, 0, stream>>>(
        x, wqkv + (size_t)li * Dd * 3 * Dd, enc_qkv_b + (size_t)li * 3 * Dd,
        qkv, Mrows, 3 * Dd, Dd, 0, vT);
    attn_kernel<<<attnGrid, 32, 0, stream>>>(qkv, vT, attn);
    gemm_at_kernel<<<dim3(Dd / 64, Mrows / 128), gemmBlk, 0, stream>>>(
        attn, wout + (size_t)li * Dd * Dd, enc_out_b + (size_t)li * Dd,
        tmp2, Mrows, Dd, Dd, 0, nullptr);
    add_ln_kernel<<<Mrows, 256, 0, stream>>>(
        x, tmp2, enc_ln1_g + (size_t)li * Dd, enc_ln1_b + (size_t)li * Dd, x);
    gemm_at_kernel<<<dim3(FFf / 64, Mrows / 128), gemmBlk, 0, stream>>>(
        x, wff1 + (size_t)li * Dd * FFf, enc_ff1_b + (size_t)li * FFf,
        hbuf, Mrows, FFf, Dd, 1, nullptr);
    gemm_at_kernel<<<dim3(Dd / 64, Mrows / 128), gemmBlk, 0, stream>>>(
        hbuf, wff2 + (size_t)li * FFf * Dd, enc_ff2_b + (size_t)li * Dd,
        tmp2, Mrows, Dd, FFf, 0, nullptr);
    add_ln_kernel<<<Mrows, 256, 0, stream>>>(
        x, tmp2, enc_ln2_g + (size_t)li * Dd, enc_ln2_b + (size_t)li * Dd, x);
  }

  // relation-aware attention (mean attn pattern == 1/L analytically)
  gemm_at_kernel<<<dim3(3 * Dd / 64, Mrows / 128), gemmBlk, 0, stream>>>(
      x, wrqkv, rel_qkv_b, qkv, Mrows, 3 * Dd, Dd, 0, vT);
  attn_kernel<<<attnGrid, 32, 0, stream>>>(qkv, vT, attn);
  gemm_at_kernel<<<dim3(Dd / 64, Mrows / 128), gemmBlk, 0, stream>>>(
      attn, wrout, rel_out_b, tmp2, Mrows, Dd, Dd, 0, nullptr);

  // logits = out @ proj_w + proj_b
  float* logits = (float*)d_out;
  gemm_at_kernel<<<dim3(Vv / 64, Mrows / 128), gemmBlk, 0, stream>>>(
      tmp2, wproj, proj_b, logits, Mrows, Vv, Dd, 0, nullptr);

  // relation_scores -> trailing B*NREL floats
  rel_scores_kernel<<<1, 32, 0, stream>>>(
      rel_cls_w, rel_cls_b, logits + (size_t)Mrows * Vv, 1.0f / (float)Ll);
}
